// HumanVAttention_46016279609762
// MI455X (gfx1250) — matrix operand order repaired and checked by reading, not compiled, not run on hardware
//
#include <hip/hip_runtime.h>
#include <hip/hip_bf16.h>

typedef _Float16 half_t;
typedef __attribute__((ext_vector_type(16))) _Float16 v16h;
typedef __attribute__((ext_vector_type(8)))  _Float16 v8h;
typedef __attribute__((ext_vector_type(8)))  float    v8f;

#define S_LEN 2048
#define HID_D 2048
#define NH    32
#define NKV   8
#define HD    64
#define KVDIM (NKV * HD)   /* 512 */
#define SCALE 0.125f       /* 64^-0.5 */

union V16U { v16h v; v8h h[2]; };

static __device__ __forceinline__ v8f wmma16(v16h a, v16h b, v8f c) {
  // D = A(16x32 f16) x B(32x16 f16) + C(16x16 f32)
  return __builtin_amdgcn_wmma_f32_16x16x32_f16(false, a, false, b, (short)0, c, false, false);
}

// Async Global -> LDS 16-byte copy (CDNA5 ASYNCcnt path, §08).
static __device__ __forceinline__ void async_ld_b128(half_t* lds, const half_t* g) {
  unsigned lo = (unsigned)(unsigned long long)lds;  // LDS byte address (low 32 bits)
  asm volatile("global_load_async_to_lds_b128 %0, %1, off" :: "v"(lo), "v"(g) : "memory");
}

// Wait until at most `n` async copies remain in flight (in-order completion).
#define WAIT_ASYNC_LE_1() asm volatile("s_wait_asynccnt 0x1" ::: "memory")

__global__ void cvt_f32_f16(const float* __restrict__ in, half_t* __restrict__ out, int n) {
  int i = blockIdx.x * blockDim.x + threadIdx.x;
  if (i < n) out[i] = (half_t)in[i];
}

// Block = 8 waves stacked on M sharing one 64-col W tile.
// W tile (64x32 halfs, 4KB) is DMA'd into LDS via global_load_async_to_lds_b128.
// Two LDS buffers, two copies in flight: before reading a buffer we wait
// asynccnt<=1 so the *older* copy (ours) has landed while the newer one still
// overlaps this step's WMMAs. Manual ping-pong (64-K super-steps) avoids
// register-rotation moves and their WMMA->VALU hazard NOPs.
// One wave computes a 32x64 tile of  Out = A(MxK) @ W^T  (W is NxK row-major).
// mode 0: store half row-major | 1: +RoPE | 2: store transposed | 3: store f32
__global__ __launch_bounds__(256) void gemm16(const half_t* __restrict__ A,
                                              const half_t* __restrict__ W,
                                              half_t* __restrict__ out_h,
                                              float* __restrict__ out_f,
                                              const float* __restrict__ cosT,
                                              const float* __restrict__ sinT,
                                              int M, int K, int N, int mode) {
  __shared__ __align__(16) half_t Bs[2][64 * 32];   // 2 x 4KB
  const int lane    = threadIdx.x & 31;
  const int wave    = threadIdx.x >> 5;
  const int lane16  = lane & 15;
  const int khalf   = lane >> 4;
  const int a0      = khalf * 16;   // A-frag K sub-offset (contiguous 16-run)
  const int b0      = khalf * 16;   // B-frag K sub-offset (contiguous 16-run)
  const int tiles_m = M >> 5;       // 32-row tiles (tiles_m % 8 == 0 here)
  const int mgroups = tiles_m >> 3;
  const int grp     = blockIdx.x % mgroups;
  const int nb      = blockIdx.x / mgroups;
  const int s0      = ((grp << 3) + wave) << 5;  // 32 rows, per wave
  const int c0      = nb << 6;                   // 64 cols, shared by block

  // per-thread slice of the B-tile DMA: one b128 each (256 * 16B = 4KB)
  const int crow = threadIdx.x >> 2;          // 0..63
  const int cchk = (threadIdx.x & 3) << 3;    // 0,8,16,24 halfs
  const half_t* gB = W + (size_t)(c0 + crow) * K + cchk;
  half_t* lB0 = &Bs[0][crow * 32 + cchk];
  half_t* lB1 = &Bs[1][crow * 32 + cchk];

  v8f acc[2][4] = {};
  const half_t* arow0 = A + (size_t)(s0 + lane16) * K;
  const half_t* arow1 = arow0 + (size_t)16 * K;

  // prologue: DMA K-steps 0 and 32 (buf0/buf1), load A frags for step 0
  async_ld_b128(lB0, gB);
  async_ld_b128(lB1, gB + 32);
  V16U afA[2], afB[2];
  afA[0].h[0] = *(const v8h*)(arow0 + a0);
  afA[0].h[1] = *(const v8h*)(arow0 + a0 + 8);
  afA[1].h[0] = *(const v8h*)(arow1 + a0);
  afA[1].h[1] = *(const v8h*)(arow1 + a0 + 8);

  for (int k0 = 0; k0 < K; k0 += 64) {
    // ---------- even 32-step: buf0 / afA ----------
    WAIT_ASYNC_LE_1();        // buf0's copy (older of the two) has landed
    __syncthreads();          // ... for every wave's slice
    {
      const int kn = k0 + 32; // prefetch A frags for the odd step
      afB[0].h[0] = *(const v8h*)(arow0 + kn + a0);
      afB[0].h[1] = *(const v8h*)(arow0 + kn + a0 + 8);
      afB[1].h[0] = *(const v8h*)(arow1 + kn + a0);
      afB[1].h[1] = *(const v8h*)(arow1 + kn + a0 + 8);
    }
    {
      const half_t* bsrc = &Bs[0][0];
      V16U bf[4];
#pragma unroll
      for (int t = 0; t < 4; ++t) {
        bf[t].h[0] = *(const v8h*)(bsrc + (t * 16 + lane16) * 32 + b0);
        bf[t].h[1] = *(const v8h*)(bsrc + (t * 16 + lane16) * 32 + b0 + 8);
      }
#pragma unroll
      for (int t = 0; t < 4; ++t) {
        acc[0][t] = wmma16(afA[0].v, bf[t].v, acc[0][t]);
        acc[1][t] = wmma16(afA[1].v, bf[t].v, acc[1][t]);
      }
    }
    __syncthreads();          // all waves done reading buf0
    if (k0 + 64 < K) async_ld_b128(lB0, gB + k0 + 64);

    // ---------- odd 32-step: buf1 / afB ----------
    WAIT_ASYNC_LE_1();        // buf1's copy has landed
    __syncthreads();
    if (k0 + 64 < K) {        // prefetch A frags for the next even step
      const int kn = k0 + 64;
      afA[0].h[0] = *(const v8h*)(arow0 + kn + a0);
      afA[0].h[1] = *(const v8h*)(arow0 + kn + a0 + 8);
      afA[1].h[0] = *(const v8h*)(arow1 + kn + a0);
      afA[1].h[1] = *(const v8h*)(arow1 + kn + a0 + 8);
    }
    {
      const half_t* bsrc = &Bs[1][0];
      V16U bf[4];
#pragma unroll
      for (int t = 0; t < 4; ++t) {
        bf[t].h[0] = *(const v8h*)(bsrc + (t * 16 + lane16) * 32 + b0);
        bf[t].h[1] = *(const v8h*)(bsrc + (t * 16 + lane16) * 32 + b0 + 8);
      }
#pragma unroll
      for (int t = 0; t < 4; ++t) {
        acc[0][t] = wmma16(afB[0].v, bf[t].v, acc[0][t]);
        acc[1][t] = wmma16(afB[1].v, bf[t].v, acc[1][t]);
      }
    }
    __syncthreads();          // all waves done reading buf1
    if (k0 + 96 < K) async_ld_b128(lB1, gB + k0 + 96);
  }

  const int m_off = khalf * 8;

  if (mode == 1) {
    // RoPE within this head: d pairs with d+32; cos[s][d] == cos[s][d+32].
#pragma unroll
    for (int rt = 0; rt < 2; ++rt)
#pragma unroll
      for (int t = 0; t < 2; ++t)
#pragma unroll
        for (int v = 0; v < 8; ++v) {
          int s = s0 + rt * 16 + v + m_off;
          int d = (t << 4) + lane16;
          float c  = cosT[s * HD + d];
          float sn = sinT[s * HD + d];
          float x1 = acc[rt][t][v], x2 = acc[rt][t + 2][v];
          acc[rt][t][v]     = x1 * c - x2 * sn;
          acc[rt][t + 2][v] = x2 * c + x1 * sn;
        }
  }

  if (mode == 2) {
#pragma unroll
    for (int rt = 0; rt < 2; ++rt)
#pragma unroll
      for (int t = 0; t < 4; ++t) {
        v8h pk;
#pragma unroll
        for (int v = 0; v < 8; ++v) pk[v] = (half_t)acc[rt][t][v];
        *(v8h*)(out_h + (size_t)(c0 + t * 16 + lane16) * M + s0 + rt * 16 + m_off) = pk;
      }
  } else if (mode == 3) {
#pragma unroll
    for (int rt = 0; rt < 2; ++rt)
#pragma unroll
      for (int t = 0; t < 4; ++t)
#pragma unroll
        for (int v = 0; v < 8; ++v)
          out_f[(size_t)(s0 + rt * 16 + v + m_off) * N + c0 + t * 16 + lane16] = acc[rt][t][v];
  } else {
#pragma unroll
    for (int rt = 0; rt < 2; ++rt)
#pragma unroll
      for (int t = 0; t < 4; ++t)
#pragma unroll
        for (int v = 0; v < 8; ++v)
          out_h[(size_t)(s0 + rt * 16 + v + m_off) * N + c0 + t * 16 + lane16] =
              (half_t)acc[rt][t][v];
  }
}

// Flash attention: one wave per (head, 16-query-row tile). Causal.
__global__ __launch_bounds__(256) void flash16(const half_t* __restrict__ Q,
                                               const half_t* __restrict__ Km,
                                               const half_t* __restrict__ Vt,
                                               half_t* __restrict__ attn) {
  __shared__ __align__(16) half_t pbuf[8 * 16 * 32];  // 1KB per wave
  const int lane   = threadIdx.x & 31;
  const int wave   = threadIdx.x >> 5;
  const int lane16 = lane & 15;
  const int khalf  = lane >> 4;
  const int a0     = khalf * 16;
  const int b0     = khalf * 16;
  const int m_off  = khalf * 8;
  const int w      = blockIdx.x * 8 + wave;
  const int qt     = w & 127;    // S/16 = 128
  const int h      = w >> 7;
  const int s0     = qt << 4;
  const int kvh    = h >> 2;     // N_REP = 4

  // Q A-fragments for d=[0,32) and d=[32,64)
  const half_t* qrow = Q + (size_t)(s0 + lane16) * (NH * HD) + h * HD;
  V16U qa0, qa1;
  qa0.h[0] = *(const v8h*)(qrow + a0);
  qa0.h[1] = *(const v8h*)(qrow + a0 + 8);
  qa1.h[0] = *(const v8h*)(qrow + 32 + a0);
  qa1.h[1] = *(const v8h*)(qrow + 32 + a0 + 8);

  v8f o[4] = {};
  float mrow[8], lrow[8];
#pragma unroll
  for (int v = 0; v < 8; ++v) { mrow[v] = -1e30f; lrow[v] = 0.f; }
  half_t* pw = pbuf + wave * (16 * 32);

  const int kend = s0 + 16;
  for (int kb = 0; kb < kend; kb += 32) {
    // prefetch next key block's K and V rows (global_prefetch_b8)
    if (kb + 32 < kend) {
      __builtin_prefetch(Km + (size_t)(kb + 32 + lane) * KVDIM + kvh * HD, 0, 1);
      __builtin_prefetch(Vt + (size_t)(kvh * HD + lane) * S_LEN + kb + 32, 0, 1);
      __builtin_prefetch(Vt + (size_t)(kvh * HD + 32 + lane) * S_LEN + kb + 32, 0, 1);
    }
    // ---- scores: S[16x32] = Q(16x64) @ K^T(64x32), two 16-col subtiles ----
    v8f st[2];
#pragma unroll
    for (int t = 0; t < 2; ++t) {
      const half_t* krow = Km + (size_t)(kb + t * 16 + lane16) * KVDIM + kvh * HD;
      V16U kf0, kf1;
      kf0.h[0] = *(const v8h*)(krow + b0);
      kf0.h[1] = *(const v8h*)(krow + b0 + 8);
      kf1.h[0] = *(const v8h*)(krow + 32 + b0);
      kf1.h[1] = *(const v8h*)(krow + 32 + b0 + 8);
      v8f c = {};
      c = wmma16(qa0.v, kf0.v, c);
      c = wmma16(qa1.v, kf1.v, c);
      st[t] = c;
    }
    // ---- online softmax (rows live in one VGPR across a 16-lane group) ----
#pragma unroll
    for (int v = 0; v < 8; ++v) {
      const int qpos = s0 + v + m_off;
      float x0 = st[0][v] * SCALE;
      float x1 = st[1][v] * SCALE;
      if (kb + lane16 > qpos)      x0 = -1e30f;
      if (kb + 16 + lane16 > qpos) x1 = -1e30f;
      float rm = fmaxf(x0, x1);
      rm = fmaxf(rm, __shfl_xor(rm, 1, 32));
      rm = fmaxf(rm, __shfl_xor(rm, 2, 32));
      rm = fmaxf(rm, __shfl_xor(rm, 4, 32));
      rm = fmaxf(rm, __shfl_xor(rm, 8, 32));
      float mn    = fmaxf(mrow[v], rm);
      float alpha = __expf(mrow[v] - mn);
      float p0 = __expf(x0 - mn);
      float p1 = __expf(x1 - mn);
      float rs = p0 + p1;
      rs += __shfl_xor(rs, 1, 32);
      rs += __shfl_xor(rs, 2, 32);
      rs += __shfl_xor(rs, 4, 32);
      rs += __shfl_xor(rs, 8, 32);
      lrow[v] = lrow[v] * alpha + rs;
      mrow[v] = mn;
#pragma unroll
      for (int c = 0; c < 4; ++c) o[c][v] = o[c][v] * alpha;
      const int m = v + m_off;
      pw[m * 32 + lane16]      = (half_t)p0;
      pw[m * 32 + 16 + lane16] = (half_t)p1;
    }
    __builtin_amdgcn_fence(__ATOMIC_ACQ_REL, "wavefront");
    // ---- P(16x32) as A-fragment from LDS ----
    V16U pa;
    pa.h[0] = *(const v8h*)(pw + lane16 * 32 + a0);
    pa.h[1] = *(const v8h*)(pw + lane16 * 32 + a0 + 8);
    // ---- O += P @ V   (V transposed: (512, S), key-contiguous) ----
#pragma unroll
    for (int c = 0; c < 4; ++c) {
      const half_t* vrow = Vt + (size_t)(kvh * HD + c * 16 + lane16) * S_LEN + kb;
      V16U vb;
      vb.h[0] = *(const v8h*)(vrow + b0);
      vb.h[1] = *(const v8h*)(vrow + b0 + 8);
      o[c] = wmma16(pa.v, vb.v, o[c]);
    }
    __builtin_amdgcn_fence(__ATOMIC_ACQ_REL, "wavefront");
  }
  // ---- finalize and store attn (S, 32*64) ----
#pragma unroll
  for (int v = 0; v < 8; ++v) {
    float inv = 1.0f / lrow[v];
    const size_t row = (size_t)(s0 + v + m_off) * (NH * HD) + h * HD;
#pragma unroll
    for (int c = 0; c < 4; ++c)
      attn[row + c * 16 + lane16] = (half_t)(o[c][v] * inv);
  }
}

extern "C" void kernel_launch(void* const* d_in, const int* in_sizes, int n_in,
                              void* d_out, int out_size, void* d_ws, size_t ws_size,
                              hipStream_t stream) {
  (void)in_sizes; (void)n_in; (void)out_size; (void)ws_size;
  const float* hs   = (const float*)d_in[0];
  const float* cosT = (const float*)d_in[1];
  const float* sinT = (const float*)d_in[2];
  // d_in[3] attention_mask unused: causal mask applied analytically
  const float* wq = (const float*)d_in[4];
  const float* wk = (const float*)d_in[5];
  const float* wv = (const float*)d_in[6];
  const float* wo = (const float*)d_in[7];
  float* out = (float*)d_out;

  half_t* ws = (half_t*)d_ws;
  const size_t SZ_H  = (size_t)S_LEN * HID_D;   // 4M elems
  const size_t SZ_WQ = (size_t)HID_D * HID_D;   // 4M
  const size_t SZ_WK = (size_t)KVDIM * HID_D;   // 1M
  half_t* Hb  = ws;
  half_t* Wqb = Hb  + SZ_H;
  half_t* Wkb = Wqb + SZ_WQ;
  half_t* Wvb = Wkb + SZ_WK;
  half_t* Wob = Wvb + SZ_WK;
  half_t* Qb  = Wob + SZ_WQ;                      // (S, 2048)
  half_t* Kb  = Qb  + SZ_H;                       // (S, 512)
  half_t* Vt  = Kb  + (size_t)S_LEN * KVDIM;      // (512, S) transposed
  half_t* At  = Vt  + (size_t)S_LEN * KVDIM;      // (S, 2048)

  cvt_f32_f16<<<(int)((SZ_H  + 255) / 256), 256, 0, stream>>>(hs, Hb,  (int)SZ_H);
  cvt_f32_f16<<<(int)((SZ_WQ + 255) / 256), 256, 0, stream>>>(wq, Wqb, (int)SZ_WQ);
  cvt_f32_f16<<<(int)((SZ_WK + 255) / 256), 256, 0, stream>>>(wk, Wkb, (int)SZ_WK);
  cvt_f32_f16<<<(int)((SZ_WK + 255) / 256), 256, 0, stream>>>(wv, Wvb, (int)SZ_WK);
  cvt_f32_f16<<<(int)((SZ_WQ + 255) / 256), 256, 0, stream>>>(wo, Wob, (int)SZ_WQ);

  // Q = H @ Wq^T (+RoPE), (S,2048): (64/8 m-groups * 32 n-tiles) = 256 blocks
  gemm16<<<(S_LEN / 32 / 8) * (HID_D / 64), 256, 0, stream>>>(
      Hb, Wqb, Qb, nullptr, cosT, sinT, S_LEN, HID_D, HID_D, 1);
  // K = H @ Wk^T (+RoPE), (S,512)
  gemm16<<<(S_LEN / 32 / 8) * (KVDIM / 64), 256, 0, stream>>>(
      Hb, Wkb, Kb, nullptr, cosT, sinT, S_LEN, HID_D, KVDIM, 1);
  // V = H @ Wv^T, stored transposed (512,S)
  gemm16<<<(S_LEN / 32 / 8) * (KVDIM / 64), 256, 0, stream>>>(
      Hb, Wvb, Vt, nullptr, nullptr, nullptr, S_LEN, HID_D, KVDIM, 2);
  // attention: 32 heads * 128 q-tiles / 8 waves = 512 blocks
  flash16<<<(NH * (S_LEN / 16)) / 8, 256, 0, stream>>>(Qb, Kb, Vt, At);
  // out = attn @ Wo^T, fp32
  gemm16<<<(S_LEN / 32 / 8) * (HID_D / 64), 256, 0, stream>>>(
      At, Wob, nullptr, out, nullptr, nullptr, S_LEN, HID_D, HID_D, 3);
}
